// MyICLSTMCell_64226940944722
// MI455X (gfx1250) — compile-verified, matrix operand order checked by reading, not run on hardware
//
#include <hip/hip_runtime.h>
#include <hip/hip_bf16.h>

#define B_ 64
#define T_ 512
#define D_ 512
#define U_ 512
#define NBLK 16  // persistent blocks in the recurrent kernel

typedef __attribute__((ext_vector_type(16))) __bf16 v16bf;
typedef __attribute__((ext_vector_type(8)))  __bf16 v8bf;
typedef __attribute__((ext_vector_type(8)))  float  v8f;
typedef __attribute__((ext_vector_type(4)))  float  v4f;

// Workspace layout (bytes)
#define XW_BYTES   ((size_t)B_ * T_ * U_ * 4)          // 64 MB f32
#define WIT_OFF    (XW_BYTES)                          // 512 KB bf16 (Wi^T)
#define UIT_OFF    (WIT_OFF + (size_t)U_ * D_ * 2)     // 512 KB bf16 (Ui^T)
#define HBUF_OFF   (UIT_OFF + (size_t)U_ * U_ * 2)     // 2 x 64 KB bf16 (h double buffer)
#define BAR_OFF    (HBUF_OFF + (size_t)2 * B_ * U_ * 2)

// ---------------------------------------------------------------------------
// Phase 0: transpose+convert weights to bf16, init h buffer 0, reset barrier.
// ---------------------------------------------------------------------------
__global__ void __launch_bounds__(256)
prep_kernel(const float* __restrict__ Wi, const float* __restrict__ Ui,
            const float* __restrict__ h0,
            __bf16* __restrict__ WiT, __bf16* __restrict__ UiT,
            __bf16* __restrict__ hbuf0, unsigned* __restrict__ bar) {
  int idx = blockIdx.x * blockDim.x + threadIdx.x;
  if (idx < U_ * D_) {
    int u = idx / D_, d = idx - u * D_;
    WiT[idx] = (__bf16)Wi[d * U_ + u];   // WiT[u][d] = Wi[d][u]
    UiT[idx] = (__bf16)Ui[d * U_ + u];   // UiT[u][k] = Ui[k][u]
  }
  if (idx < B_ * U_) hbuf0[idx] = (__bf16)h0[idx];
  if (idx == 0) bar[0] = 0u;
}

// ---------------------------------------------------------------------------
// Phase 1: xW[bt][u] = x[bt][:] . Wi[:][u]   (bf16 WMMA, f32 accumulate)
// One wave per 16x16 tile; 2048 M-tiles x 32 N-tiles = 65536 waves.
// ---------------------------------------------------------------------------
__global__ void __launch_bounds__(256)
xproj_kernel(const float* __restrict__ x, const __bf16* __restrict__ WiT,
             float* __restrict__ xW) {
  const int lane = threadIdx.x & 31;
  const int wave = threadIdx.x >> 5;
  const int w = blockIdx.x * 8 + wave;       // 0..65535
  const int ntile = w & 31;
  const int mtile = w >> 5;                  // 0..2047
  const int m0 = mtile * 16, n0 = ntile * 16;
  const int half = lane >> 4;                // 0 | 1
  const int l15 = lane & 15;
  const int lo = half * 8;                   // K split between lane halves

  const float*  arow = x   + (long)(m0 + l15) * D_;
  const __bf16* brow = WiT + (long)(n0 + l15) * D_;

  v8f acc = {};
#pragma unroll
  for (int kk = 0; kk < 16; ++kk) {
    const int kb = kk * 32;
    // A operand: load f32, convert to bf16 in registers
    v4f a0 = *(const v4f*)(arow + kb + lo);
    v4f a1 = *(const v4f*)(arow + kb + lo + 4);
    v4f a2 = *(const v4f*)(arow + kb + lo + 16);
    v4f a3 = *(const v4f*)(arow + kb + lo + 20);
    v16bf a;
#pragma unroll
    for (int e = 0; e < 4; ++e) {
      a[e]      = (__bf16)a0[e];
      a[e + 4]  = (__bf16)a1[e];
      a[e + 8]  = (__bf16)a2[e];
      a[e + 12] = (__bf16)a3[e];
    }
    // B operand: pre-transposed bf16, contiguous in K
    v8bf b0 = *(const v8bf*)(brow + kb + lo);
    v8bf b1 = *(const v8bf*)(brow + kb + lo + 16);
    v16bf b;
#pragma unroll
    for (int e = 0; e < 8; ++e) { b[e] = b0[e]; b[e + 8] = b1[e]; }

    acc = __builtin_amdgcn_wmma_f32_16x16x32_bf16(
        false, a, false, b, (short)0, acc, false, false);
  }
#pragma unroll
  for (int r = 0; r < 8; ++r) {
    int row = m0 + r + half * 8;             // bt index
    xW[(long)row * U_ + n0 + l15] = acc[r];
  }
}

// ---------------------------------------------------------------------------
// Phase 2: persistent recurrent scan. 16 blocks x 8 waves = 128 waves,
// one 16x16 (batch x unit) tile per wave. B-operand (UiT slice) is hoisted
// into 128 VGPRs and reused for all 512 timesteps. Grid barrier via a
// monotonic global atomic counter.
// ---------------------------------------------------------------------------
__global__ void __launch_bounds__(256)
recur_kernel(const float* __restrict__ xW, const __bf16* __restrict__ UiT,
             const float* __restrict__ c0,
             const float* __restrict__ DWi, const float* __restrict__ bi,
             const float* __restrict__ DWf, const float* __restrict__ bfv,
             const float* __restrict__ DWo, const float* __restrict__ bo,
             const float* __restrict__ DWc, const float* __restrict__ bc,
             __bf16* __restrict__ hbuf, float* __restrict__ out,
             unsigned* __restrict__ bar) {
  const int lane = threadIdx.x & 31;
  const int wave = threadIdx.x >> 5;
  const int w = blockIdx.x * 8 + wave;       // 0..127
  const int mt = w & 3;                      // 4 batch tiles (64 rows)
  const int nt = w >> 2;                     // 32 unit tiles (512 cols)
  const int m0 = mt * 16, n0 = nt * 16;
  const int half = lane >> 4, l15 = lane & 15;
  const int lo = half * 8;
  const int col = n0 + l15;

  // Per-column diagonal gate parameters: constant per lane for entire scan
  const float dwi = DWi[col], dbi = bi[col];
  const float dwf = DWf[col], dbf = bfv[col];
  const float dwo = DWo[col], dbo = bo[col];
  const float dwc = DWc[col], dbc = bc[col];

  float cst[8];
  int   rows[8];
#pragma unroll
  for (int r = 0; r < 8; ++r) {
    rows[r] = m0 + r + half * 8;             // batch index
    cst[r]  = c0[rows[r] * U_ + col];
  }

  // Hoist the time-invariant B operand (UiT[col][0..511]) into registers:
  // 16 k-steps x 16 bf16 = 512 B/lane = 128 VGPRs.
  const __bf16* brow = UiT + (long)col * U_;
  v16bf bmat[16];
#pragma unroll
  for (int kk = 0; kk < 16; ++kk) {
    v8bf b0 = *(const v8bf*)(brow + kk * 32 + lo);
    v8bf b1 = *(const v8bf*)(brow + kk * 32 + lo + 16);
#pragma unroll
    for (int e = 0; e < 8; ++e) { bmat[kk][e] = b0[e]; bmat[kk][e + 8] = b1[e]; }
  }

  for (int t = 0; t < T_; ++t) {
    const __bf16* hb = hbuf + (size_t)(t & 1) * (B_ * U_);
    __bf16*       hn = hbuf + (size_t)((t + 1) & 1) * (B_ * U_);

    // C init = xW[:, t, :] tile
    v8f acc;
#pragma unroll
    for (int r = 0; r < 8; ++r)
      acc[r] = xW[((long)rows[r] * T_ + t) * U_ + col];
    if (t + 1 < T_)  // warm L2/L0 for next step's xW slice
      __builtin_prefetch(&xW[((long)rows[0] * T_ + t + 1) * U_ + col], 0, 1);

    const __bf16* arow = hb + (long)(m0 + l15) * U_;
#pragma unroll
    for (int kk = 0; kk < 16; ++kk) {
      const int kb = kk * 32;
      v8bf a0 = *(const v8bf*)(arow + kb + lo);
      v8bf a1 = *(const v8bf*)(arow + kb + lo + 16);
      v16bf a;
#pragma unroll
      for (int e = 0; e < 8; ++e) { a[e] = a0[e]; a[e + 8] = a1[e]; }
      acc = __builtin_amdgcn_wmma_f32_16x16x32_bf16(
          false, a, false, bmat[kk], (short)0, acc, false, false);
    }

    // Diagonal gates + state update, write h (f32 to out, bf16 to next buffer)
#pragma unroll
    for (int r = 0; r < 8; ++r) {
      float z  = acc[r];
      float ig = fmaxf(fmaf(dwi, z, dbi), 0.f);
      float fg = fmaxf(fmaf(dwf, z, dbf), 0.f);
      float og = fmaxf(fmaf(dwo, z, dbo), 0.f);
      float gg = fmaxf(fmaf(dwc, z, dbc), 0.f);
      float nc = fg * cst[r] + ig * gg;
      cst[r] = nc;
      float nh = og * fmaxf(nc, 0.f);
      out[((long)rows[r] * T_ + t) * U_ + col] = nh;
      hn[rows[r] * U_ + col] = (__bf16)nh;
    }

    // ---- grid-wide barrier (monotonic counter, 16 arrivals/step) ----
    __threadfence();                 // make hn stores visible device-wide
    __syncthreads();
    if (threadIdx.x == 0) {
      atomicAdd(bar, 1u);
      const unsigned target = (unsigned)(NBLK * (t + 1));
      while (__hip_atomic_load(bar, __ATOMIC_ACQUIRE,
                               __HIP_MEMORY_SCOPE_AGENT) < target)
        __builtin_amdgcn_s_sleep(1);
    }
    __syncthreads();
    __threadfence();                 // invalidate L0 before reading new h
  }
}

// ---------------------------------------------------------------------------
extern "C" void kernel_launch(void* const* d_in, const int* in_sizes, int n_in,
                              void* d_out, int out_size, void* d_ws, size_t ws_size,
                              hipStream_t stream) {
  const float* x   = (const float*)d_in[0];
  const float* h0  = (const float*)d_in[1];
  const float* c0  = (const float*)d_in[2];
  const float* Wi  = (const float*)d_in[3];
  const float* Ui  = (const float*)d_in[4];
  const float* DWi = (const float*)d_in[5];
  const float* bi  = (const float*)d_in[6];
  const float* DWf = (const float*)d_in[7];
  const float* bf  = (const float*)d_in[8];
  const float* DWo = (const float*)d_in[9];
  const float* bo  = (const float*)d_in[10];
  const float* DWc = (const float*)d_in[11];
  const float* bc  = (const float*)d_in[12];

  char* ws = (char*)d_ws;
  float*    xW   = (float*)(ws);
  __bf16*   WiT  = (__bf16*)(ws + WIT_OFF);
  __bf16*   UiT  = (__bf16*)(ws + UIT_OFF);
  __bf16*   hbuf = (__bf16*)(ws + HBUF_OFF);
  unsigned* bar  = (unsigned*)(ws + BAR_OFF);
  float*    out  = (float*)d_out;

  prep_kernel<<<(U_ * D_ + 255) / 256, 256, 0, stream>>>(Wi, Ui, h0, WiT, UiT,
                                                         hbuf, bar);
  // 65536 wave-tiles / 8 waves per block
  xproj_kernel<<<8192, 256, 0, stream>>>(x, WiT, xW);
  // persistent scan: exactly NBLK co-resident blocks
  recur_kernel<<<NBLK, 256, 0, stream>>>(xW, UiT, c0, DWi, bi, DWf, bf, DWo,
                                         bo, DWc, bc, hbuf, out, bar);
}
